// FFT3_77472620085499
// MI455X (gfx1250) — matrix operand order repaired and checked by reading, not compiled, not run on hardware
//
#include <hip/hip_runtime.h>
#include <hip/hip_bf16.h>
#include <math.h>

typedef __bf16 bf16;
typedef __attribute__((ext_vector_type(16))) __bf16 v16bf;
typedef __attribute__((ext_vector_type(8)))  float  v8f;
typedef __attribute__((ext_vector_type(4)))  unsigned int v4u;
typedef __attribute__((ext_vector_type(8)))  unsigned int v8u;

#define BATCH 8
#define DIM   64
#define CH    256
#define HH    256
#define WW    256
#define HWSZ  (HH*WW)            // 65536
#define NPATCH (BATCH*32*32)     // 8192 patches total
#define NPIX  (BATCH*HWSZ)       // 524288 pixels total
#define PADW  262                // 256 + 2*3 halo, zero border for SAME dwconv
#define PLANE (PADW*PADW)        // 68644 elements per (c,b) padded plane

__device__ __forceinline__ v8f wmma_bf16(v16bf a, v16bf b, v8f c) {
  // D = A(16x32 bf16) * B(32x16 bf16) + C(16x16 f32)
  return __builtin_amdgcn_wmma_f32_16x16x32_bf16(
      false, a, false, b, (short)0, c, false, false);
}

// ---------------- K0a: convert 1x1 conv weights to bf16 -------------------
__global__ void k_prep_weights(const float* __restrict__ w_in,
                               const float* __restrict__ w_out,
                               bf16* __restrict__ w_in_bf,
                               bf16* __restrict__ w_out_bf) {
  int i = blockIdx.x * 256 + threadIdx.x;
  if (i < CH * DIM)  w_in_bf[i]  = (bf16)w_in[i];   // (C,D) flat
  if (i < DIM * CH)  w_out_bf[i] = (bf16)w_out[i];  // (D,C) flat
}

// ---------------- K0b: fft_filter -> circulant matrix M_c (64x64) ---------
// Real multiplier in rfft2 space == circular conv with k_c = irfft2(F_c).
__global__ void k_build_circulant(const float* __restrict__ fft_filter,
                                  bf16* __restrict__ Mbf) {
  __shared__ float ksp[64];
  const int c = blockIdx.x;          // 256 blocks
  const int t = threadIdx.x;         // 64 threads
  const float cs[8] = {1.f, 0.70710678f, 0.f, -0.70710678f,
                       -1.f, -0.70710678f, 0.f, 0.70710678f};
  const int a = t >> 3, b = t & 7;
  const float* F = fft_filter + c * 40;     // 8 x 5 half-spectrum
  float acc = 0.f;
  for (int u = 0; u < 8; ++u)
    for (int v = 0; v < 8; ++v) {
      float g = (v <= 4) ? F[u * 5 + v] : F[((8 - u) & 7) * 5 + (8 - v)];
      acc += g * cs[(u * a + v * b) & 7];   // Hermitian ext., imag cancels
    }
  ksp[t] = acc * (1.f / 64.f);
  __syncthreads();
  const int io = t >> 3, jo = t & 7;
  bf16* Mrow = Mbf + (size_t)c * 4096 + t * 64;
  for (int pi = 0; pi < 64; ++pi) {
    int ii = pi >> 3, ji = pi & 7;
    Mrow[pi] = (bf16)ksp[((io - ii) & 7) * 8 + ((jo - ji) & 7)];
  }
}

// ---------------- K0c: zero the 3-pixel border of every padded plane ------
__global__ void k_zero_border(bf16* __restrict__ yp2) {
  bf16* plane = yp2 + (size_t)blockIdx.x * PLANE;   // 2048 planes
  const int t = threadIdx.x;
  for (int i = t; i < 3108; i += 256) {
    int row, col;
    if (i < 786)       { row = i / PADW;                 col = i % PADW; }
    else if (i < 1572) { row = 259 + (i - 786) / PADW;   col = (i - 786) % PADW; }
    else { int j = i - 1572; row = 3 + j / 6; int k = j % 6; col = (k < 3) ? k : 256 + k; }
    plane[row * PADW + col] = (bf16)0.f;
  }
}

// ---------------- K1: conv_in 1x1 as WMMA GEMM ----------------------------
__global__ void k_conv_in(const float* __restrict__ x,
                          const bf16* __restrict__ w_in_bf,
                          const float* __restrict__ b_in,
                          bf16* __restrict__ y1) {
  const int wave = (blockIdx.x * blockDim.x + threadIdx.x) >> 5; // 32768 waves
  const int lane = threadIdx.x & 31;
  const int b  = wave >> 12;
  const int nt = wave & 4095;
  const int h  = nt >> 4;
  const int w0 = (nt & 15) << 4;
  const int nIdx = lane & 15;
  const int half = lane >> 4;

  const float* xb = x + ((size_t)b * DIM) * HWSZ + h * WW + w0 + nIdx;
  v16bf bfrag[2];
  for (int kk = 0; kk < 2; ++kk)
    for (int e = 0; e < 16; ++e) {
      int k = kk * 32 + e + 16 * half;
      bfrag[kk][e] = (bf16)xb[(size_t)k * HWSZ];
    }

  const int wpix  = w0 + nIdx;
  const int p     = ((h & 7) << 3) | (wpix & 7);
  const int patch = b * 1024 + ((h >> 3) << 5) + (wpix >> 3);

  for (int mt = 0; mt < 16; ++mt) {
    const int m0 = mt * 16;
    v8f acc = {};
    const bf16* Arow = w_in_bf + (size_t)(m0 + nIdx) * DIM;
    for (int kk = 0; kk < 2; ++kk) {
      v16bf a;
      for (int e = 0; e < 16; ++e) {
        int k = kk * 32 + e + 16 * half;
        a[e] = Arow[k];
      }
      acc = wmma_bf16(a, bfrag[kk], acc);
    }
    for (int v = 0; v < 8; ++v) {
      int ch = m0 + v + 8 * half;
      float val = acc[v] + b_in[ch];
      y1[((size_t)ch * 64 + p) * NPATCH + patch] = (bf16)val;
    }
  }
}

// ---------------- K2: spectral filter as per-channel circulant GEMM -------
// writes into zero-padded planes: yp2[c][b][(h+3)][(w+3)]
__global__ void k_spectral(const bf16* __restrict__ y1,
                           const bf16* __restrict__ Mbf,
                           bf16* __restrict__ yp2) {
  const int wave = (blockIdx.x * blockDim.x + threadIdx.x) >> 5; // 131072 waves
  const int lane = threadIdx.x & 31;
  const int c  = wave >> 9;
  const int n0 = (wave & 511) << 4;
  const int nIdx = lane & 15;
  const int half = lane >> 4;

  const bf16* Bc = y1 + (size_t)c * 64 * NPATCH;
  v16bf bfrag[2];
  for (int kk = 0; kk < 2; ++kk)
    for (int e = 0; e < 16; ++e) {
      int k = kk * 32 + e + 16 * half;
      bfrag[kk][e] = Bc[(size_t)k * NPATCH + n0 + nIdx];
    }

  const bf16* Ac = Mbf + (size_t)c * 4096;
  const int n = n0 + nIdx;
  const int b = n >> 10, pidx = n & 1023;
  const int ph = pidx >> 5, pw = pidx & 31;
  bf16* plane = yp2 + ((size_t)c * BATCH + b) * PLANE;

  for (int mt = 0; mt < 4; ++mt) {
    const int m0 = mt * 16;
    v8f acc = {};
    const bf16* Arow = Ac + (size_t)(m0 + nIdx) * 64;
    for (int kk = 0; kk < 2; ++kk) {
      v16bf a;
      for (int e = 0; e < 16; ++e) {
        int k = kk * 32 + e + 16 * half;
        a[e] = Arow[k];
      }
      acc = wmma_bf16(a, bfrag[kk], acc);
    }
    for (int v = 0; v < 8; ++v) {
      int pout = m0 + v + 8 * half;
      int hh = ph * 8 + (pout >> 3), ww = pw * 8 + (pout & 7);
      plane[(hh + 3) * PADW + (ww + 3)] = (bf16)acc[v];
    }
  }
}

// ---------------- K3a: TDM-staged depthwise 7x7 + bias + exact GELU -------
// Wave 0 issues a Tensor Data Mover load of the 22x40 halo tile into LDS,
// waits on TENSORcnt, then all 8 waves consume it. Tile start is always
// in-bounds (padded plane); right-edge overreach is zero-filled by the
// descriptor's tensor_dim OOB rule.
__global__ void k_dwconv_gelu(const bf16* __restrict__ yp2,
                              const float* __restrict__ w_dw,
                              const float* __restrict__ b_dw,
                              bf16* __restrict__ g) {
  __shared__ struct { bf16 tile[22 * 40]; float wk[49]; } sm;  // tile at LDS addr 0
  const int blk = blockIdx.x;                 // (c*8+b)*128 + tileid
  const int tileid = blk & 127;               // 16 (h) x 8 (w) tiles
  const int cb = blk >> 7;
  const int b = cb & 7;
  const int c = cb >> 3;
  const int ty0 = (tileid >> 3) * 16;         // 16 rows per tile
  const int tx0 = (tileid & 7) * 32;          // 32 cols per tile
  const int t = threadIdx.x;                  // 256 threads

  if (t < 49) sm.wk[t] = w_dw[c * 49 + t];

  if (t < 32) {                               // wave 0 only: issue TDM
    const bf16* src = yp2 + ((size_t)c * BATCH + b) * PLANE
                          + (size_t)ty0 * PADW + tx0;
    unsigned long long ga = (unsigned long long)src;
    v4u g0;
    g0[0] = 1u;                               // count=1, no gather
    g0[1] = 0u;                               // lds_addr = 0 (sm.tile)
    g0[2] = (unsigned int)ga;                 // global_addr[31:0]
    g0[3] = (unsigned int)((ga >> 32) & 0x01FFFFFFu) | (2u << 30); // type=2
    v8u g1;
    g1[0] = 1u << 16;                         // data_size=1 (2 bytes)
    g1[1] = (unsigned int)PADW << 16;         // tensor_dim0 = 262
    g1[2] = (unsigned int)PADW << 16;         // tensor_dim1 = 262
    g1[3] = 40u << 16;                        // tile_dim0 = 40 elems (80 B)
    g1[4] = 22u;                              // tile_dim1 = 22 rows
    g1[5] = (unsigned int)PADW;               // tensor_dim0_stride = 262
    g1[6] = 0u;
    g1[7] = 0u;
    asm volatile("tensor_load_to_lds %0, %1" : : "s"(g0), "s"(g1) : "memory");
    __builtin_amdgcn_s_wait_tensorcnt(0);
  }
  __syncthreads();

  const int tx = t & 15, ty = t >> 4;         // each thread: 2 pixels (tx, tx+16)
  float aX = b_dw[c], aY = aX;
  for (int ky = 0; ky < 7; ++ky) {
    const bf16* row = &sm.tile[(ty + ky) * 40 + tx];
    for (int kx = 0; kx < 7; ++kx) {
      float w = sm.wk[ky * 7 + kx];
      aX = fmaf(w, (float)row[kx], aX);
      aY = fmaf(w, (float)row[kx + 16], aY);
    }
  }
  float gX = 0.5f * aX * (1.f + erff(aX * 0.70710678118f));
  float gY = 0.5f * aY * (1.f + erff(aY * 0.70710678118f));
  size_t qbase = (size_t)c * NPIX + (size_t)b * HWSZ + (ty0 + ty) * WW + tx0 + tx;
  g[qbase]      = (bf16)gX;
  g[qbase + 16] = (bf16)gY;
}

// ---------------- K3b: conv_out 1x1 as WMMA GEMM + bias -------------------
__global__ void k_conv_out(const bf16* __restrict__ g,
                           const bf16* __restrict__ w_out_bf,
                           const float* __restrict__ b_out,
                           float* __restrict__ out) {
  const int wave = (blockIdx.x * blockDim.x + threadIdx.x) >> 5; // 32768 waves
  const int lane = threadIdx.x & 31;
  const int q0 = wave << 4;
  const int nIdx = lane & 15;
  const int half = lane >> 4;

  v8f zero = {};
  v8f acc[4] = {zero, zero, zero, zero};
  for (int kk = 0; kk < 8; ++kk) {
    v16bf bfrag;
    for (int e = 0; e < 16; ++e) {
      int cidx = kk * 32 + e + 16 * half;
      bfrag[e] = g[(size_t)cidx * NPIX + q0 + nIdx];
    }
    for (int mt = 0; mt < 4; ++mt) {
      v16bf a;
      const bf16* Arow = w_out_bf + (size_t)(mt * 16 + nIdx) * CH;
      for (int e = 0; e < 16; ++e) {
        int k = kk * 32 + e + 16 * half;
        a[e] = Arow[k];
      }
      acc[mt] = wmma_bf16(a, bfrag, acc[mt]);
    }
  }
  const int q = q0 + nIdx;
  const int b = q >> 16, hw = q & 65535;
  for (int mt = 0; mt < 4; ++mt)
    for (int v = 0; v < 8; ++v) {
      int d = mt * 16 + v + 8 * half;
      out[((size_t)b * DIM + d) * HWSZ + hw] = acc[mt][v] + b_out[d];
    }
}

extern "C" void kernel_launch(void* const* d_in, const int* in_sizes, int n_in,
                              void* d_out, int out_size, void* d_ws, size_t ws_size,
                              hipStream_t stream) {
  const float* x          = (const float*)d_in[0];
  const float* w_in       = (const float*)d_in[1];
  const float* b_in       = (const float*)d_in[2];
  const float* fft_filter = (const float*)d_in[3];
  const float* w_dw       = (const float*)d_in[4];
  const float* b_dw       = (const float*)d_in[5];
  const float* w_out      = (const float*)d_in[6];
  const float* b_out      = (const float*)d_in[7];
  float* out = (float*)d_out;

  char* ws = (char*)d_ws;
  bf16* w_in_bf  = (bf16*)ws;                                   // 32 KB
  bf16* w_out_bf = (bf16*)(ws + (32 << 10));                    // 32 KB
  bf16* Mbf      = (bf16*)(ws + (64 << 10));                    // 2 MB
  bf16* bufA     = (bf16*)(ws + (64 << 10) + (2 << 20));        // 256 MB: y1, then g
  bf16* bufB     = (bf16*)(ws + (64 << 10) + (2 << 20) + ((size_t)256 << 20)); // ~269 MB padded y2

  k_prep_weights   <<<64,     256, 0, stream>>>(w_in, w_out, w_in_bf, w_out_bf);
  k_build_circulant<<<CH,      64, 0, stream>>>(fft_filter, Mbf);
  k_zero_border    <<<2048,   256, 0, stream>>>(bufB);
  k_conv_in        <<<4096,   256, 0, stream>>>(x, w_in_bf, b_in, bufA);
  k_spectral       <<<16384,  256, 0, stream>>>(bufA, Mbf, bufB);
  k_dwconv_gelu    <<<262144, 256, 0, stream>>>(bufB, w_dw, b_dw, bufA);
  k_conv_out       <<<4096,   256, 0, stream>>>(bufA, w_out_bf, b_out, out);
}
